// TransformerBlock_46084999086296
// MI455X (gfx1250) — matrix operand order repaired and checked by reading, not compiled, not run on hardware
//
#include <hip/hip_runtime.h>
#include <hip/hip_bf16.h>
#include <stdint.h>

// ---------------------------------------------------------------------------
// TransformerBlock on gfx1250 (MI455X).
// All GEMMs: v_wmma_f32_16x16x32_bf16, LDS double-buffered tiles staged with
// global_load_async_to_lds_b128 (ASYNCcnt) — CDNA5 async copy path.
// ---------------------------------------------------------------------------

typedef __attribute__((ext_vector_type(16))) __bf16 v16bf;
typedef __attribute__((ext_vector_type(8)))  __bf16 v8bf;
typedef __attribute__((ext_vector_type(8)))  float  v8f;

#define N_CTX    4096
#define D_MODEL  2048
#define D_HIDDEN 8192

// Flip to 0 if the async-to-LDS mnemonic is rejected by the assembler.
#define USE_ASYNC_LDS 1

__device__ __forceinline__ __bf16 f2bf(float f) {
  union { float f; uint32_t u; } v; v.f = f;
  uint32_t r = v.u + 0x7FFFu + ((v.u >> 16) & 1u);   // round-to-nearest-even
  uint16_t h = (uint16_t)(r >> 16);
  return __builtin_bit_cast(__bf16, h);
}

// ---------------------------------------------------------------------------
// Async global->LDS copy of 16B per lane (ISA: GLOBAL_LOAD_ASYNC_TO_LDS_B128,
// each lane copies 16B from its VADDR address to the LDS address in VDST, tracked by ASYNCcnt).
// Generic LDS pointer low 32 bits == wave-relative LDS byte address.
// ---------------------------------------------------------------------------
__device__ __forceinline__ void async_copy16(__bf16* lds, const __bf16* g) {
#if USE_ASYNC_LDS
  unsigned lds_off = (unsigned)(uintptr_t)lds;
  asm volatile("global_load_async_to_lds_b128 %0, %1, off"
               :: "v"(lds_off), "v"(g) : "memory");
#else
  *(v8bf*)lds = *(const v8bf*)g;     // sync fallback: global_load + ds_store
#endif
}

#if USE_ASYNC_LDS
#define WAIT_ASYNC_LE_8() asm volatile("s_wait_asynccnt 0x8" ::: "memory")
#define WAIT_ASYNC_LE_0() asm volatile("s_wait_asynccnt 0x0" ::: "memory")
#else
#define WAIT_ASYNC_LE_8()
#define WAIT_ASYNC_LE_0()
#endif

// ---------------------------------------------------------------------------
// Fragment loaders (LDS -> VGPR); A and B use the same contiguous k order, k = 16*(l/16) + elem.
// A (16x32 bf16 MxK): lane l holds row (l&15);
//   lanes 0-15:  elems 0-15 = K 0..15
//   lanes 16-31: elems 0-15 = K 16..31
// B (32x16 bf16) from NxK-major tile: lane l holds col (l&15);
//   lanes 0-15 hold K 0..15, lanes 16-31 hold K 16..31 (contiguous)
// ---------------------------------------------------------------------------
__device__ __forceinline__ v16bf lds_frag_a(const __bf16* p) {
  v8bf lo = *(const v8bf*)(p);
  v8bf hi = *(const v8bf*)(p + 8);
  v16bf r;
#pragma unroll
  for (int i = 0; i < 8; ++i) { r[i] = lo[i]; r[i + 8] = hi[i]; }
  return r;
}
__device__ __forceinline__ v16bf lds_frag_b(const __bf16* p) {
  return *(const v16bf*)(p);
}

// Epilogue modes
#define MODE_F32        0   // Cf = acc
#define MODE_BIAS_RELU  1   // Cb = bf16(relu(acc + bias[n]))
#define MODE_BF16       2   // Cb = bf16(acc)
#define MODE_MASK       3   // Cf = (col<=row) ? acc : -3e38   (causal scores)
#define MODE_FINAL      4   // Cf = res1 + res2 + relu(acc + bias[n])

// Tiling: block 128x128, BK=64 per stage, 8 waves, wave tile 32x64 (2x4 WMMA).
// LDS: A/B tiles 128 rows x 64 elems, row stride padded to 72 elems (144B) so
// 16-lane fans (stride 144B -> bank step 36 mod 64) are conflict-free; 16B
// aligned for ds_load_b128. Double buffered: 2*2*128*72*2B = 72KB (<320KB WGP).
#define BK       64
#define LDS_STR  72
#define TILE_ELS (128 * LDS_STR)

// C[M,N] = A[MxK] (bf16 row-major) * Bt[NxK] (bf16 row-major, i.e. B^T)
template <int MODE>
__global__ __launch_bounds__(256)
void gemm_bf16_wmma(const __bf16* __restrict__ A, const __bf16* __restrict__ Bt,
                    int M, int N, int K,
                    const float* __restrict__ bias,
                    const float* __restrict__ res1,
                    const float* __restrict__ res2,
                    float* __restrict__ Cf, __bf16* __restrict__ Cb) {
  __shared__ __bf16 smem[2][2][TILE_ELS];   // [buf][A=0/B=1][...]

  const int t    = threadIdx.x;
  const int lane = t & 31;
  const int wave = t >> 5;
  const int wm = wave & 3;                  // 4 waves along M
  const int wn = wave >> 2;                 // 2 waves along N
  const int m0 = blockIdx.y * 128;
  const int n0 = blockIdx.x * 128;

  const __bf16* Ag = A  + (size_t)m0 * K;   // tile top-left (k added per stage)
  const __bf16* Bg = Bt + (size_t)n0 * K;

  // Stage one BK-slice of A and B: 128 rows x 64 elems = 1024 16B-chunks each;
  // 256 threads x 4 chunks; each wave issues 8 async copies per stage.
  auto stage = [&](int buf, int k) {
#pragma unroll
    for (int r = 0; r < 4; ++r) {
      const int c   = t + r * 256;          // chunk id 0..1023
      const int row = c >> 3;               // 0..127
      const int off = (c & 7) * 8;          // elem offset 0..56
      async_copy16(&smem[buf][0][row * LDS_STR + off], Ag + (size_t)row * K + k + off);
      async_copy16(&smem[buf][1][row * LDS_STR + off], Bg + (size_t)row * K + k + off);
    }
  };

  const int arow  = lane & 15;
  const int akofs = (lane < 16) ? 0 : 16;
  const int bcol  = lane & 15;
  const int bkofs = (lane < 16) ? 0 : 16;

  v8f acc[2][4] = {};

  const int nStages = K / BK;
  stage(0, 0);
  int buf = 0;
  for (int s = 0; s < nStages; ++s) {
    if (s + 1 < nStages) {
      stage(buf ^ 1, (s + 1) * BK);
      WAIT_ASYNC_LE_8();                    // current buffer's 8 copies done
    } else {
      WAIT_ASYNC_LE_0();
    }
    __syncthreads();                        // staged data visible to all waves

    const __bf16* sa = smem[buf][0];
    const __bf16* sb = smem[buf][1];
#pragma unroll
    for (int ks = 0; ks < BK; ks += 32) {
      v16bf a[2], b[4];
#pragma unroll
      for (int i = 0; i < 2; ++i)
        a[i] = lds_frag_a(sa + (wm * 32 + i * 16 + arow) * LDS_STR + ks + akofs);
#pragma unroll
      for (int j = 0; j < 4; ++j)
        b[j] = lds_frag_b(sb + (wn * 64 + j * 16 + bcol) * LDS_STR + ks + bkofs);
#pragma unroll
      for (int i = 0; i < 2; ++i)
#pragma unroll
        for (int j = 0; j < 4; ++j)
          acc[i][j] = __builtin_amdgcn_wmma_f32_16x16x32_bf16(
              false, a[i], false, b[j], (short)0, acc[i][j], false, false);
    }

    __syncthreads();                        // done reading buf before reuse
    buf ^= 1;
  }

  // C/D f32 layout: lane l holds col (l&15); VGPR v = row v (lanes 0-15) or
  // row v+8 (lanes 16-31).
  const int ccol = lane & 15;
  const int crow = (lane < 16) ? 0 : 8;
#pragma unroll
  for (int i = 0; i < 2; ++i) {
#pragma unroll
    for (int j = 0; j < 4; ++j) {
      const int gm = m0 + wm * 32 + i * 16 + crow;
      const int gn = n0 + wn * 64 + j * 16 + ccol;
#pragma unroll
      for (int v = 0; v < 8; ++v) {
        const float val = acc[i][j][v];
        const int row = gm + v;
        const int col = gn;
        const size_t idx = (size_t)row * N + col;
        if constexpr (MODE == MODE_F32) {
          Cf[idx] = val;
        } else if constexpr (MODE == MODE_BIAS_RELU) {
          Cb[idx] = f2bf(fmaxf(val + bias[col], 0.0f));
        } else if constexpr (MODE == MODE_BF16) {
          Cb[idx] = f2bf(val);
        } else if constexpr (MODE == MODE_MASK) {
          Cf[idx] = (col <= row) ? val : -3.0e38f;
        } else { // MODE_FINAL
          Cf[idx] = res1[idx] + res2[idx] + fmaxf(val + bias[col], 0.0f);
        }
      }
    }
  }
}

// ---------------------------------------------------------------------------
// f32 -> bf16 conversion (grid-stride)
// ---------------------------------------------------------------------------
__global__ void cvt_f32_bf16(const float* __restrict__ in, __bf16* __restrict__ out, size_t n) {
  size_t i = (size_t)blockIdx.x * blockDim.x + threadIdx.x;
  const size_t stride = (size_t)gridDim.x * blockDim.x;
  for (; i < n; i += stride) out[i] = f2bf(in[i]);
}

// f32[rows x cols] -> bf16[cols x rows]  (LDS 32x32 tile, bank-conflict padded)
__global__ void cvt_transpose_bf16(const float* __restrict__ in, __bf16* __restrict__ out,
                                   int rows, int cols) {
  __shared__ __bf16 tile[32][33];
  const int tx = threadIdx.x, ty = threadIdx.y;
  const int c  = blockIdx.x * 32 + tx;
  const int r0 = blockIdx.y * 32;
#pragma unroll
  for (int i = 0; i < 32; i += 8)
    tile[ty + i][tx] = f2bf(in[(size_t)(r0 + ty + i) * cols + c]);
  __syncthreads();
  const int rr  = blockIdx.y * 32 + tx;   // output minor index (original row)
  const int cc0 = blockIdx.x * 32;        // output major base (original col)
#pragma unroll
  for (int i = 0; i < 32; i += 8)
    out[(size_t)(cc0 + ty + i) * rows + rr] = tile[tx][ty + i];
}

// ---------------------------------------------------------------------------
// Row softmax over masked scores, writing bf16 probs (one block per row)
// ---------------------------------------------------------------------------
__global__ __launch_bounds__(256)
void softmax_rows(const float* __restrict__ scores, __bf16* __restrict__ probs, int n) {
  const int row = blockIdx.x;
  const float* s = scores + (size_t)row * n;
  __bf16* p = probs + (size_t)row * n;
  __shared__ float red[256];

  float mx = -3.4e38f;
  for (int i = threadIdx.x; i < n; i += 256) mx = fmaxf(mx, s[i]);
  red[threadIdx.x] = mx; __syncthreads();
  for (int o = 128; o > 0; o >>= 1) {
    if (threadIdx.x < o) red[threadIdx.x] = fmaxf(red[threadIdx.x], red[threadIdx.x + o]);
    __syncthreads();
  }
  mx = red[0]; __syncthreads();

  float sum = 0.0f;
  for (int i = threadIdx.x; i < n; i += 256) sum += __expf(s[i] - mx);
  red[threadIdx.x] = sum; __syncthreads();
  for (int o = 128; o > 0; o >>= 1) {
    if (threadIdx.x < o) red[threadIdx.x] += red[threadIdx.x + o];
    __syncthreads();
  }
  const float inv = 1.0f / red[0];

  for (int i = threadIdx.x; i < n; i += 256) p[i] = f2bf(__expf(s[i] - mx) * inv);
}

// ---------------------------------------------------------------------------
extern "C" void kernel_launch(void* const* d_in, const int* in_sizes, int n_in,
                              void* d_out, int out_size, void* d_ws, size_t ws_size,
                              hipStream_t stream) {
  (void)in_sizes; (void)n_in; (void)out_size; (void)ws_size;
  const float* x    = (const float*)d_in[0];
  const float* qk   = (const float*)d_in[1];
  const float* ov   = (const float*)d_in[2];
  const float* wup  = (const float*)d_in[3];
  const float* bup  = (const float*)d_in[4];
  const float* whid = (const float*)d_in[5];
  const float* bhid = (const float*)d_in[6];
  const float* wdn  = (const float*)d_in[7];
  const float* bdn  = (const float*)d_in[8];
  float* out = (float*)d_out;

  char* ws = (char*)d_ws;
  const size_t MB = 1024u * 1024u;
  __bf16* xb     = (__bf16*)(ws +   0 * MB);  // [N_CTX,   D_MODEL ] 16MB
  __bf16* xT     = (__bf16*)(ws +  16 * MB);  // [D_MODEL, N_CTX   ] 16MB
  __bf16* qkT    = (__bf16*)(ws +  32 * MB);  // [D_MODEL, D_MODEL ]  8MB
  __bf16* ovT    = (__bf16*)(ws +  40 * MB);  // [D_MODEL, D_MODEL ]  8MB
  __bf16* wupb   = (__bf16*)(ws +  48 * MB);  // [D_HIDDEN,D_MODEL ] 32MB (already NxK)
  __bf16* whidb  = (__bf16*)(ws +  80 * MB);  // [D_HIDDEN,D_HIDDEN]128MB (already NxK)
  __bf16* wdnb   = (__bf16*)(ws + 208 * MB);  // [D_MODEL, D_HIDDEN] 32MB (already NxK)
  __bf16* qb     = (__bf16*)(ws + 240 * MB);  // [N_CTX,   D_MODEL ] 16MB
  float*  scores = (float* )(ws + 256 * MB);  // [N_CTX,   N_CTX   ] 64MB
  __bf16* probs  = (__bf16*)(ws + 320 * MB);  // [N_CTX,   N_CTX   ] 32MB
  __bf16* a0b    = (__bf16*)(ws + 352 * MB);  // [N_CTX,   D_MODEL ] 16MB
  float*  attnf  = (float* )(ws + 368 * MB);  // [N_CTX,   D_MODEL ] 32MB
  __bf16* h1b    = (__bf16*)(ws + 400 * MB);  // [N_CTX,   D_HIDDEN] 64MB
  __bf16* h2b    = (__bf16*)(ws + 464 * MB);  // [N_CTX,   D_HIDDEN] 64MB
  // total workspace: 528MB

  // --- bf16 conversions (w.T weights are already NxK for the WMMA B side) ---
  cvt_f32_bf16<<<4096, 256, 0, stream>>>(x,    xb,    (size_t)N_CTX * D_MODEL);
  cvt_f32_bf16<<<4096, 256, 0, stream>>>(wup,  wupb,  (size_t)D_HIDDEN * D_MODEL);
  cvt_f32_bf16<<<8192, 256, 0, stream>>>(whid, whidb, (size_t)D_HIDDEN * D_HIDDEN);
  cvt_f32_bf16<<<4096, 256, 0, stream>>>(wdn,  wdnb,  (size_t)D_MODEL * D_HIDDEN);
  cvt_transpose_bf16<<<dim3(D_MODEL/32, N_CTX/32),   dim3(32, 8), 0, stream>>>(x,  xT,  N_CTX,   D_MODEL);
  cvt_transpose_bf16<<<dim3(D_MODEL/32, D_MODEL/32), dim3(32, 8), 0, stream>>>(qk, qkT, D_MODEL, D_MODEL);
  cvt_transpose_bf16<<<dim3(D_MODEL/32, D_MODEL/32), dim3(32, 8), 0, stream>>>(ov, ovT, D_MODEL, D_MODEL);

  // --- attention ---
  gemm_bf16_wmma<MODE_BF16><<<dim3(D_MODEL/128, N_CTX/128), 256, 0, stream>>>(
      xb, qkT, N_CTX, D_MODEL, D_MODEL, nullptr, nullptr, nullptr, nullptr, qb);
  gemm_bf16_wmma<MODE_MASK><<<dim3(N_CTX/128, N_CTX/128), 256, 0, stream>>>(
      qb, xb, N_CTX, N_CTX, D_MODEL, nullptr, nullptr, nullptr, scores, nullptr);
  softmax_rows<<<N_CTX, 256, 0, stream>>>(scores, probs, N_CTX);
  gemm_bf16_wmma<MODE_BF16><<<dim3(D_MODEL/128, N_CTX/128), 256, 0, stream>>>(
      probs, xT, N_CTX, D_MODEL, N_CTX, nullptr, nullptr, nullptr, nullptr, a0b);
  gemm_bf16_wmma<MODE_F32><<<dim3(D_MODEL/128, N_CTX/128), 256, 0, stream>>>(
      a0b, ovT, N_CTX, D_MODEL, D_MODEL, nullptr, nullptr, nullptr, attnf, nullptr);

  // --- MLP ---
  gemm_bf16_wmma<MODE_BIAS_RELU><<<dim3(D_HIDDEN/128, N_CTX/128), 256, 0, stream>>>(
      xb, wupb, N_CTX, D_HIDDEN, D_MODEL, bup, nullptr, nullptr, nullptr, h1b);
  gemm_bf16_wmma<MODE_BIAS_RELU><<<dim3(D_HIDDEN/128, N_CTX/128), 256, 0, stream>>>(
      h1b, whidb, N_CTX, D_HIDDEN, D_HIDDEN, bhid, nullptr, nullptr, nullptr, h2b);
  gemm_bf16_wmma<MODE_FINAL><<<dim3(D_MODEL/128, N_CTX/128), 256, 0, stream>>>(
      h2b, wdnb, N_CTX, D_MODEL, D_HIDDEN, bdn, x, attnf, out, nullptr);
}
